// Edge_Encoder_Residual_41291815584028
// MI455X (gfx1250) — matrix operand order repaired and checked by reading, not compiled, not run on hardware
//
#include <hip/hip_runtime.h>
#include <hip/hip_bf16.h>
#include <cstdint>

// ---------------- problem constants (match reference) ----------------
#define NN        100000      // total nodes (B * NUM_EDGES)
#define NEDGE     50000       // nodes per sample
#define EG        400000      // graph edges before self loops
#define ETOT      (EG + NN)   // 500000
#define F_IN      16
#define TDIM      64
#define HC        128
#define NHEAD     4
#define CH        32
#define KP0       96          // F_IN+TDIM=80 padded to multiple of 32
#define THETA_F   1.0f
#define NEG_SLOPE 0.2f

typedef __attribute__((ext_vector_type(16))) __bf16 v16bf;
typedef __attribute__((ext_vector_type(8)))  float  v8f;

union ABFrag { uint4 q[2]; v16bf v; };

__device__ __forceinline__ unsigned short f2bf(float f) {
    unsigned u = __float_as_uint(f);
    unsigned r = (u + 0x7FFFu + ((u >> 16) & 1u)) >> 16;   // round-to-nearest-even
    return (unsigned short)r;
}

// ---------------- tiny time-embedding MLP (B=2, 64 wide) ----------------
__global__ void time_embed_kernel(const int* __restrict__ t,
                                  const float* __restrict__ tW0, const float* __restrict__ tb0,
                                  const float* __restrict__ tW1, const float* __restrict__ tb1,
                                  float* __restrict__ te /* [2][64] */) {
    __shared__ float e0[2][TDIM];
    __shared__ float e1[2][TDIM];
    int j = threadIdx.x;                     // 0..63
    const float c = -logf(10000.0f) / 31.0f; // half-1 = 31
    for (int b = 0; b < 2; ++b) {
        float ts = (float)t[b];              // * (1000/1000) = 1
        float v;
        if (j < 32) { float f = __expf(c * j);        v = __sinf(ts * f); }
        else        { float f = __expf(c * (j - 32)); v = __cosf(ts * f); }
        e0[b][j] = v;
    }
    __syncthreads();
    for (int b = 0; b < 2; ++b) {
        float acc = tb0[j];
        for (int k = 0; k < TDIM; ++k) acc += e0[b][k] * tW0[k * TDIM + j];
        e1[b][j] = acc / (1.0f + __expf(-acc));   // silu
    }
    __syncthreads();
    for (int b = 0; b < 2; ++b) {
        float acc = tb1[j];
        for (int k = 0; k < TDIM; ++k) acc += e1[b][k] * tW1[k * TDIM + j];
        te[b * TDIM + j] = acc / (1.0f + __expf(-acc));
    }
}

// ---------------- build h0 = [x | te | pad] as bf16, stride KP0 ----------------
__global__ void concat_h0_kernel(const float* __restrict__ x, const float* __restrict__ te,
                                 unsigned short* __restrict__ hbf) {
    size_t i = (size_t)blockIdx.x * blockDim.x + threadIdx.x;
    if (i >= (size_t)NN * KP0) return;
    int n = (int)(i / KP0), j = (int)(i % KP0);
    float v;
    if (j < F_IN)             v = x[(size_t)n * F_IN + j];
    else if (j < F_IN + TDIM) v = te[(n / NEDGE) * TDIM + (j - F_IN)];
    else                      v = 0.0f;
    hbf[i] = f2bf(v);
}

// ---------------- pack weight [K,Nout] f32 -> WMMA B-fragment bf16 ----------------
// layout: [k0/32][tile_n][lane 0..31][16 bf16]; lane<16 holds K=k0..k0+15 of col
// (tile_n*16 + lane), lane>=16 holds K=k0+16..k0+31 (contiguous k order, same as the A fragment loads).
__global__ void pack_w_kernel(const float* __restrict__ W, unsigned short* __restrict__ out,
                              int K, int Kp, int Nout) {
    size_t idx = (size_t)blockIdx.x * blockDim.x + threadIdx.x;
    size_t total = (size_t)Kp * Nout;
    if (idx >= total) return;
    int tilesN = Nout >> 4;
    int i    = (int)(idx & 15);
    int lane = (int)((idx >> 4) & 31);
    size_t rest = idx >> 9;
    int tn = (int)(rest % tilesN);
    int kc = (int)(rest / tilesN);
    int half = lane >> 4;
    int col  = tn * 16 + (lane & 15);
    int k    = kc * 32 + half * 16 + i;
    float v = (k < K) ? W[(size_t)k * Nout + col] : 0.0f;
    out[idx] = f2bf(v);
}

// ---------------- WMMA GEMM with A-reuse + LDS-staged B ----------------
// One wave produces a full 16-row x Nout strip: each A fragment is loaded once
// from global and reused across all TILESN WMMA calls; packed B fragments are
// staged into LDS once per block and shared by all 4 waves (ds_load_b128).
template <int KCH, int TILESN>
__global__ void __launch_bounds__(128)
wmma_gemm_tiled(const unsigned short* __restrict__ Xbf,
                const unsigned short* __restrict__ Wpk,
                const float* __restrict__ bias,
                float* __restrict__ Y, int act) {
    constexpr int Kp   = KCH * 32;
    constexpr int Nout = TILESN * 16;
    constexpr int BQ   = KCH * TILESN * 64;          // B size in uint4 (16B) units
    __shared__ uint4 sB[BQ];                         // <=32 KB of 320 KB WGP LDS

    // cooperative stage of all B fragments (coalesced b128 loads)
    for (int q = threadIdx.x; q < BQ; q += blockDim.x)
        sB[q] = ((const uint4*)Wpk)[q];
    __syncthreads();

    int wv   = threadIdx.x >> 5;
    int lane = threadIdx.x & 31;
    int tile_m = blockIdx.x * (blockDim.x >> 5) + wv;
    if (tile_m >= NN / 16) return;                   // uniform per wave
    int half = lane >> 4;
    int row  = tile_m * 16 + (lane & 15);
    const unsigned short* xrow = Xbf + (size_t)row * Kp;
    const unsigned short* sBs  = (const unsigned short*)sB;

    v8f acc[TILESN];
#pragma unroll
    for (int tn = 0; tn < TILESN; ++tn) acc[tn] = {};

#pragma unroll
    for (int kc = 0; kc < KCH; ++kc) {
        ABFrag a;
        // A fragment: lane<16 holds K k0..k0+15 of its row,
        // lane>=16 holds K k0+16..k0+31.
        const unsigned short* pa = xrow + kc * 32 + half * 16;
        a.q[0] = *(const uint4*)(pa);
        a.q[1] = *(const uint4*)(pa + 8);
#pragma unroll
        for (int tn = 0; tn < TILESN; ++tn) {
            ABFrag b;
            const unsigned short* pb = sBs + (kc * TILESN + tn) * 512 + lane * 16;
            b.q[0] = *(const uint4*)(pb);
            b.q[1] = *(const uint4*)(pb + 8);
            acc[tn] = __builtin_amdgcn_wmma_f32_16x16x32_bf16(
                          false, a.v, false, b.v, (short)0, acc[tn], false, false);
        }
    }

#pragma unroll
    for (int tn = 0; tn < TILESN; ++tn) {
        int ncol = tn * 16 + (lane & 15);
        float bv = bias ? bias[ncol] : 0.0f;
#pragma unroll
        for (int r = 0; r < 8; ++r) {                // C layout: VGPR r -> M=r (+8 hi half)
            int m = tile_m * 16 + r + half * 8;
            float v = acc[tn][r] + bv;
            if (act) v = v > 0.0f ? v : 0.0f;
            Y[(size_t)m * Nout + ncol] = v;
        }
    }
}

// ---------------- per-layer attention state init ----------------
__global__ void init_attn_kernel(float* __restrict__ outacc, float* __restrict__ denom,
                                 unsigned* __restrict__ mEnc) {
    size_t i = (size_t)blockIdx.x * blockDim.x + threadIdx.x;
    if (i < (size_t)NN * HC) outacc[i] = 0.0f;
    if (i < (size_t)NN * NHEAD) { denom[i] = 0.0f; mEnc[i] = 0u; }
}

// order-preserving float<->uint for atomic max
__device__ __forceinline__ unsigned encf(float x) {
    unsigned u = __float_as_uint(x);
    return (u & 0x80000000u) ? ~u : (u | 0x80000000u);
}
__device__ __forceinline__ float decf(unsigned e) {
    return __uint_as_float((e & 0x80000000u) ? (e & 0x7FFFFFFFu) : ~e);
}

// ---------------- edge pass 1: scores + segment max (wave per edge) ----------------
__global__ void edge_score_kernel(const float* __restrict__ XL, const float* __restrict__ XR,
                                  const float* __restrict__ att, const int* __restrict__ ei,
                                  float* __restrict__ scores, unsigned* __restrict__ mEnc) {
    int wid  = blockIdx.x * (blockDim.x >> 5) + (threadIdx.x >> 5);
    int lane = threadIdx.x & 31;
    if (wid >= ETOT) return;
    int s, d;
    if (wid < EG) { s = ei[wid]; d = ei[EG + wid]; } else { s = d = wid - EG; }
    const float* xls = XL + (size_t)s * HC;
    const float* xrd = XR + (size_t)d * HC;
#pragma unroll
    for (int h = 0; h < NHEAD; ++h) {
        float g = xls[h * CH + lane] + xrd[h * CH + lane];
        g = g > 0.0f ? g : NEG_SLOPE * g;                     // leaky_relu
        float p = g * att[h * CH + lane];
        for (int off = 16; off; off >>= 1) p += __shfl_xor(p, off, 32);
        if (lane == 0) {
            scores[(size_t)wid * NHEAD + h] = p;
            atomicMax(&mEnc[(size_t)d * NHEAD + h], encf(p));
        }
    }
}

// ---------------- edge pass 2: exp, denom, un-normalized weighted scatter ----------------
__global__ void edge_accum_kernel(const float* __restrict__ XL, const float* __restrict__ scores,
                                  const unsigned* __restrict__ mEnc, const int* __restrict__ ei,
                                  float* __restrict__ denom, float* __restrict__ outacc) {
    int wid  = blockIdx.x * (blockDim.x >> 5) + (threadIdx.x >> 5);
    int lane = threadIdx.x & 31;
    if (wid >= ETOT) return;
    int s, d;
    if (wid < EG) { s = ei[wid]; d = ei[EG + wid]; } else { s = d = wid - EG; }
    float ex[NHEAD];
#pragma unroll
    for (int h = 0; h < NHEAD; ++h) {
        float m = decf(mEnc[(size_t)d * NHEAD + h]);
        ex[h] = __expf(scores[(size_t)wid * NHEAD + h] - m);
    }
    if (lane < NHEAD) atomicAdd(&denom[(size_t)d * NHEAD + lane], ex[lane]);
    const float* xls = XL + (size_t)s * HC;
    float* od = outacc + (size_t)d * HC;
#pragma unroll
    for (int h = 0; h < NHEAD; ++h)
        atomicAdd(&od[h * CH + lane], ex[h] * xls[h * CH + lane]);
}

// ---------------- node finalize: softmax divide, relu, +residual, -> bf16 ----------------
__global__ void node_finalize_kernel(const float* __restrict__ outacc, const float* __restrict__ denom,
                                     const float* __restrict__ res, const float* __restrict__ bias,
                                     unsigned short* __restrict__ hbf) {
    size_t i = (size_t)blockIdx.x * blockDim.x + threadIdx.x;
    if (i >= (size_t)NN * HC) return;
    int n = (int)(i >> 7), j = (int)(i & 127);
    float v = outacc[i] / denom[(size_t)n * NHEAD + (j >> 5)];
    if (bias) v += bias[j];
    v = v > 0.0f ? v : 0.0f;
    hbf[i] = f2bf(THETA_F * v + res[i]);          // res already relu'd in GEMM
}

// ---------------- final 32 -> 2 classifier ----------------
__global__ void logits_kernel(const float* __restrict__ fd, const float* __restrict__ acW,
                              const float* __restrict__ acb, float* __restrict__ out) {
    int n = blockIdx.x * blockDim.x + threadIdx.x;
    if (n >= NN) return;
    float a0 = acb[0], a1 = acb[1];
    const float* f = fd + (size_t)n * CH;
#pragma unroll 8
    for (int c = 0; c < CH; ++c) {
        float v = f[c];
        a0 += v * acW[c * 2 + 0];
        a1 += v * acW[c * 2 + 1];
    }
    out[n * 2 + 0] = a0;
    out[n * 2 + 1] = a1;
}

// =====================================================================
extern "C" void kernel_launch(void* const* d_in, const int* in_sizes, int n_in,
                              void* d_out, int out_size, void* d_ws, size_t ws_size,
                              hipStream_t stream) {
    // inputs (setup_inputs order)
    const float* x       = (const float*)d_in[0];
    const int*   ei      = (const int*)  d_in[1];
    const int*   t       = (const int*)  d_in[2];
    const float* tW0     = (const float*)d_in[3];
    const float* tb0     = (const float*)d_in[4];
    const float* tW1     = (const float*)d_in[5];
    const float* tb1     = (const float*)d_in[6];
    const float* c0_Wl   = (const float*)d_in[7];
    const float* c0_bl   = (const float*)d_in[8];
    const float* c0_Wr   = (const float*)d_in[9];
    const float* c0_br   = (const float*)d_in[10];
    const float* c0_att  = (const float*)d_in[11];
    const float* c0_bias = (const float*)d_in[12];
    const float* r0_W    = (const float*)d_in[13];
    const float* r0_b    = (const float*)d_in[14];
    const float* cWl     = (const float*)d_in[15];
    const float* cWr     = (const float*)d_in[16];
    const float* catt    = (const float*)d_in[17];
    const float* rW      = (const float*)d_in[18];
    const float* rb      = (const float*)d_in[19];
    const float* fdW     = (const float*)d_in[20];
    const float* fdb     = (const float*)d_in[21];
    const float* acW     = (const float*)d_in[22];
    const float* acb     = (const float*)d_in[23];
    float* out = (float*)d_out;

    // workspace carve-out (256B aligned)
    char* w = (char*)d_ws;
    auto carve = [&](size_t bytes) { char* p = w; w += (bytes + 255) & ~(size_t)255; return p; };
    float*          te     = (float*)         carve(2 * TDIM * sizeof(float));
    unsigned short* hbf    = (unsigned short*)carve((size_t)NN * HC * 2);   // max stride 128
    float*          XL     = (float*)         carve((size_t)NN * HC * 4);
    float*          XR     = (float*)         carve((size_t)NN * HC * 4);
    float*          RES    = (float*)         carve((size_t)NN * HC * 4);
    float*          OUTA   = (float*)         carve((size_t)NN * HC * 4);
    unsigned*       mEnc   = (unsigned*)      carve((size_t)NN * NHEAD * 4);
    float*          denom  = (float*)         carve((size_t)NN * NHEAD * 4);
    float*          scores = (float*)         carve((size_t)ETOT * NHEAD * 4);
    float*          fd     = (float*)         carve((size_t)NN * CH * 4);
    unsigned short* pk_c0l = (unsigned short*)carve((size_t)KP0 * HC * 2);
    unsigned short* pk_c0r = (unsigned short*)carve((size_t)KP0 * HC * 2);
    unsigned short* pk_r0  = (unsigned short*)carve((size_t)KP0 * HC * 2);
    unsigned short* pk_cl[3]; unsigned short* pk_cr[3]; unsigned short* pk_r[3];
    for (int i = 0; i < 3; ++i) {
        pk_cl[i] = (unsigned short*)carve((size_t)HC * HC * 2);
        pk_cr[i] = (unsigned short*)carve((size_t)HC * HC * 2);
        pk_r[i]  = (unsigned short*)carve((size_t)HC * HC * 2);
    }
    unsigned short* pk_fd = (unsigned short*)carve((size_t)HC * CH * 2);

    const int TPB = 256;
    auto cdiv = [](long long a, long long b) { return (int)((a + b - 1) / b); };

    // 1) time embedding MLP
    time_embed_kernel<<<1, 64, 0, stream>>>(t, tW0, tb0, tW1, tb1, te);

    // 2) h0 = [x | te] (bf16, stride KP0)
    concat_h0_kernel<<<cdiv((long long)NN * KP0, TPB), TPB, 0, stream>>>(x, te, hbf);

    // 3) pack all weights into WMMA B-fragment layout
    pack_w_kernel<<<cdiv(KP0 * HC, TPB), TPB, 0, stream>>>(c0_Wl, pk_c0l, 80, KP0, HC);
    pack_w_kernel<<<cdiv(KP0 * HC, TPB), TPB, 0, stream>>>(c0_Wr, pk_c0r, 80, KP0, HC);
    pack_w_kernel<<<cdiv(KP0 * HC, TPB), TPB, 0, stream>>>(r0_W,  pk_r0,  80, KP0, HC);
    for (int i = 0; i < 3; ++i) {
        pack_w_kernel<<<cdiv(HC * HC, TPB), TPB, 0, stream>>>(cWl + (size_t)i * HC * HC, pk_cl[i], HC, HC, HC);
        pack_w_kernel<<<cdiv(HC * HC, TPB), TPB, 0, stream>>>(cWr + (size_t)i * HC * HC, pk_cr[i], HC, HC, HC);
        pack_w_kernel<<<cdiv(HC * HC, TPB), TPB, 0, stream>>>(rW  + (size_t)i * HC * HC, pk_r[i],  HC, HC, HC);
    }
    pack_w_kernel<<<cdiv(HC * CH, TPB), TPB, 0, stream>>>(fdW, pk_fd, HC, HC, CH);

    // GEMM launch shape: 4 waves/block, one 16-row x Nout strip per wave
    const int tilesM    = NN / 16;                    // 6250
    const int gemmGrid  = cdiv(tilesM, 4);            // 1563
    const int edgeBlocks = cdiv(ETOT, 4);             // 4 waves (edges) per 128-thread block

    // ---- 4 GAT layers ----
    for (int layer = 0; layer < 4; ++layer) {
        const unsigned short* pWl; const unsigned short* pWr; const unsigned short* pR;
        const float *bl, *br, *rbias, *attp, *gbias;
        if (layer == 0) {
            pWl = pk_c0l; pWr = pk_c0r; pR = pk_r0;
            bl = c0_bl; br = c0_br; rbias = r0_b; attp = c0_att; gbias = c0_bias;
            wmma_gemm_tiled<3, 8><<<gemmGrid, 128, 0, stream>>>(hbf, pR,  rbias, RES, 1);
            wmma_gemm_tiled<3, 8><<<gemmGrid, 128, 0, stream>>>(hbf, pWl, bl,    XL,  0);
            wmma_gemm_tiled<3, 8><<<gemmGrid, 128, 0, stream>>>(hbf, pWr, br,    XR,  0);
        } else {
            int i = layer - 1;
            pWl = pk_cl[i]; pWr = pk_cr[i]; pR = pk_r[i];
            rbias = rb + (size_t)i * HC;
            attp = catt + (size_t)i * NHEAD * CH; gbias = nullptr;
            wmma_gemm_tiled<4, 8><<<gemmGrid, 128, 0, stream>>>(hbf, pR,  rbias,   RES, 1);
            wmma_gemm_tiled<4, 8><<<gemmGrid, 128, 0, stream>>>(hbf, pWl, nullptr, XL,  0);
            wmma_gemm_tiled<4, 8><<<gemmGrid, 128, 0, stream>>>(hbf, pWr, nullptr, XR,  0);
        }
        // segment softmax + message aggregation
        init_attn_kernel<<<cdiv((long long)NN * HC, TPB), TPB, 0, stream>>>(OUTA, denom, mEnc);
        edge_score_kernel<<<edgeBlocks, 128, 0, stream>>>(XL, XR, attp, ei, scores, mEnc);
        edge_accum_kernel<<<edgeBlocks, 128, 0, stream>>>(XL, scores, mEnc, ei, denom, OUTA);
        node_finalize_kernel<<<cdiv((long long)NN * HC, TPB), TPB, 0, stream>>>(OUTA, denom, RES, gbias, hbf);
    }

    // ---- head: fd = h @ fdW + fdb (WMMA), then 32->2 logits ----
    wmma_gemm_tiled<4, 2><<<gemmGrid, 128, 0, stream>>>(hbf, pk_fd, fdb, fd, 0);
    logits_kernel<<<cdiv(NN, TPB), TPB, 0, stream>>>(fd, acW, acb, out);
}
